// CausalSelfAttention_66357244723589
// MI455X (gfx1250) — matrix operand order repaired and checked by reading, not compiled, not run on hardware
//
#include <hip/hip_runtime.h>
#include <stdint.h>

// ---------------------------------------------------------------------------
// CDNA5 (gfx1250) causal GQA attention block:
//   K1: qkv = x @ qkv_w^T + qkv_b          (f32 in -> f16 out, WMMA f16/f32acc)
//   K2: flash-style attention w/ analytic ALiBi + causal mask (WMMA + async LDS)
//   K3: out = attn @ proj_w^T + proj_b     (f16 in -> f32 out, WMMA)
// GEMMs use 128x128 block tiles, 32x64 wave tiles (8 WMMAs per K-step per wave)
// and register double-buffering of the global->LDS pipeline.
// ---------------------------------------------------------------------------

typedef __attribute__((ext_vector_type(16))) _Float16 v16h;
typedef __attribute__((ext_vector_type(8)))  float    v8f;

union FragH { v16h v; uint32_t u[8]; _Float16 h[16]; };
union FragF { v8f  v; float    f[8]; };

__device__ __forceinline__ uint32_t pack_h2(_Float16 a, _Float16 b) {
  union { _Float16 h[2]; uint32_t u; } t;
  t.h[0] = a; t.h[1] = b;
  return t.u;
}
__device__ __forceinline__ uint32_t pack_f2(float a, float b) {
  return pack_h2((_Float16)a, (_Float16)b);
}

// A fragment: 16x32 f16, tile stored row-major [m][k] in LDS with 'stride' f16.
// Lane l: m = mo + (l&15), hi = l>>4 ; reg r holds K = 2r + hi*16 (+1)
__device__ __forceinline__ void load_a(FragH& f, const _Float16* base, int stride, int mo) {
  const int lane = threadIdx.x & 31;
  const int m = mo + (lane & 15);
  const int hi = lane >> 4;
#pragma unroll
  for (int r = 0; r < 8; ++r) {
    const int k = (r << 1) + (hi << 4);
    f.u[r] = *(const uint32_t*)(base + (size_t)m * stride + k);
  }
}

// B fragment: 32x16 f16 (KxN), source tile stored as [n][k] row-major in LDS.
// Lane l: n = no + (l&15), hi = l>>4 ; reg r holds K = 2r + hi*16 (+1)
__device__ __forceinline__ void load_b(FragH& f, const _Float16* base, int stride, int no) {
  const int lane = threadIdx.x & 31;
  const int n = no + (lane & 15);
  const int hi = lane >> 4;
#pragma unroll
  for (int r = 0; r < 8; ++r) {
    const int k = (r << 1) + (hi << 4);
    f.u[r] = *(const uint32_t*)(base + (size_t)n * stride + k);
  }
}

// B fragment from a [k][n] row-major tile (V matrix): gather 2 rows per reg.
__device__ __forceinline__ void load_bv(FragH& f, const _Float16* base, int stride,
                                        int kbase, int no) {
  const int lane = threadIdx.x & 31;
  const int n = no + (lane & 15);
  const int hi = lane >> 4;
#pragma unroll
  for (int r = 0; r < 8; ++r) {
    const int k = kbase + (r << 1) + (hi << 4);
    f.u[r] = pack_h2(base[(size_t)k * stride + n], base[(size_t)(k + 1) * stride + n]);
  }
}

__device__ __forceinline__ v8f wmma16(const FragH& a, const FragH& b, v8f c) {
  return __builtin_amdgcn_wmma_f32_16x16x32_f16(false, a.v, false, b.v,
                                                (short)0, c, false, false);
}

// ---------------------------------------------------------------------------
// K1: qkvh[4096][3072] (f16) = x[4096][2048] @ qkv_w[3072][2048]^T + qkv_b
// Block tile 128x128, wave tile 32x64, register double-buffered K pipeline.
// ---------------------------------------------------------------------------
__global__ __launch_bounds__(256, 1)
void k_qkv_gemm(const float* __restrict__ X, const float* __restrict__ W,
                const float* __restrict__ BIAS, _Float16* __restrict__ OUT) {
  constexpr int TM = 128, TN = 128, TK = 32, LDT = TK + 2;
  __shared__ _Float16 As[TM][LDT];
  __shared__ _Float16 Bs[TN][LDT];

  const int tid = threadIdx.x;
  const int wid = tid >> 5;
  const int lane = tid & 31;
  const int bm = blockIdx.x & 31;          // 32 M-blocks
  const int bn = blockIdx.x >> 5;          // 24 N-blocks
  const int m0 = bm * TM, n0 = bn * TN;
  const int wm = (wid & 3) * 32;           // 4 m-tiles of 32
  const int wn = (wid >> 2) * 64;          // 2 n-tiles of 64

  FragF acc[2][4];
#pragma unroll
  for (int i = 0; i < 2; ++i)
#pragma unroll
    for (int j = 0; j < 4; ++j)
#pragma unroll
      for (int r = 0; r < 8; ++r) acc[i][j].f[r] = 0.0f;

  // per-thread global slots: A and B each 128 rows x 8 float4 = 1024 / 256 = 4
  float4 pa[4], pb[4];
#pragma unroll
  for (int i = 0; i < 4; ++i) {
    const int s = tid + (i << 8);
    const int row = s >> 3, kp = (s & 7) << 2;
    pa[i] = *(const float4*)(X + (size_t)(m0 + row) * 2048 + kp);
    pb[i] = *(const float4*)(W + (size_t)(n0 + row) * 2048 + kp);
  }
#pragma unroll
  for (int i = 0; i < 4; ++i) {
    const int s = tid + (i << 8);
    const int row = s >> 3, kp = (s & 7) << 2;
    *(uint32_t*)&As[row][kp]     = pack_f2(pa[i].x, pa[i].y);
    *(uint32_t*)&As[row][kp + 2] = pack_f2(pa[i].z, pa[i].w);
    *(uint32_t*)&Bs[row][kp]     = pack_f2(pb[i].x, pb[i].y);
    *(uint32_t*)&Bs[row][kp + 2] = pack_f2(pb[i].z, pb[i].w);
  }
  __syncthreads();

  for (int k0 = TK; k0 <= 2048; k0 += TK) {
    const bool has_next = (k0 < 2048);
    if (has_next) {
#pragma unroll
      for (int i = 0; i < 4; ++i) {
        const int s = tid + (i << 8);
        const int row = s >> 3, kp = (s & 7) << 2;
        pa[i] = *(const float4*)(X + (size_t)(m0 + row) * 2048 + k0 + kp);
        pb[i] = *(const float4*)(W + (size_t)(n0 + row) * 2048 + k0 + kp);
      }
    }
    // compute on current LDS tile: 2 A frags x 4 B frags = 8 WMMAs
    FragH a0, a1, b[4];
    load_a(a0, &As[0][0], LDT, wm);
    load_a(a1, &As[0][0], LDT, wm + 16);
#pragma unroll
    for (int j = 0; j < 4; ++j) load_b(b[j], &Bs[0][0], LDT, wn + (j << 4));
#pragma unroll
    for (int j = 0; j < 4; ++j) {
      acc[0][j].v = wmma16(a0, b[j], acc[0][j].v);
      acc[1][j].v = wmma16(a1, b[j], acc[1][j].v);
    }
    if (has_next) {
      __syncthreads();
#pragma unroll
      for (int i = 0; i < 4; ++i) {
        const int s = tid + (i << 8);
        const int row = s >> 3, kp = (s & 7) << 2;
        *(uint32_t*)&As[row][kp]     = pack_f2(pa[i].x, pa[i].y);
        *(uint32_t*)&As[row][kp + 2] = pack_f2(pa[i].z, pa[i].w);
        *(uint32_t*)&Bs[row][kp]     = pack_f2(pb[i].x, pb[i].y);
        *(uint32_t*)&Bs[row][kp + 2] = pack_f2(pb[i].z, pb[i].w);
      }
      __syncthreads();
    }
  }

#pragma unroll
  for (int i = 0; i < 2; ++i)
#pragma unroll
    for (int j = 0; j < 4; ++j)
#pragma unroll
      for (int r = 0; r < 8; ++r) {
        const int ml = wm + (i << 4) + ((lane >> 4) << 3) + r;
        const int nl = wn + (j << 4) + (lane & 15);
        const float v = acc[i][j].f[r] + BIAS[n0 + nl];
        OUT[(size_t)(m0 + ml) * 3072 + (n0 + nl)] = (_Float16)v;
      }
}

// ---------------------------------------------------------------------------
// K2: attention. Grid: [32 q-tiles][16 heads][2 batch]; block = 256 = 8 waves.
// ---------------------------------------------------------------------------
__global__ __launch_bounds__(256, 1)
void k_attn(const _Float16* __restrict__ QKV, _Float16* __restrict__ OF) {
  __shared__ _Float16 Qs[64][130];
  __shared__ _Float16 Ks[64][130];
  __shared__ __align__(16) unsigned char SVbuf[64 * 130 * 2];  // Sf (f32 64x65) aliases Vs
  float    (*Sf)[65]  = (float(*)[65])SVbuf;
  _Float16 (*Vs)[130] = (_Float16(*)[130])SVbuf;
  __shared__ _Float16 Ps[64][66];
  __shared__ float row_m[64], row_l[64], row_scale[64];

  const int tid  = threadIdx.x;
  const int lane = tid & 31;
  const int wid  = tid >> 5;
  const int it = blockIdx.x & 31;
  const int h  = (blockIdx.x >> 5) & 15;
  const int bb = blockIdx.x >> 9;
  const int kvh = h >> 2;
  const int qcol = h * 128;
  const int kcol = 2048 + kvh * 128;
  const int vcol = 2560 + kvh * 128;
  const int row0 = bb * 2048 + it * 64;

  // Load Q tile (64 x 128 f16)
#pragma unroll
  for (int i = 0; i < 4; ++i) {
    const int s = tid + (i << 8);
    const int r = s >> 4, c = (s & 15) << 3;
    const uint4 d = *(const uint4*)(QKV + (size_t)(row0 + r) * 3072 + qcol + c);
    *(uint32_t*)&Qs[r][c]     = d.x;
    *(uint32_t*)&Qs[r][c + 2] = d.y;
    *(uint32_t*)&Qs[r][c + 4] = d.z;
    *(uint32_t*)&Qs[r][c + 6] = d.w;
  }
  if (tid < 64) { row_m[tid] = -1e30f; row_l[tid] = 0.0f; row_scale[tid] = 0.0f; }

  FragF accO[4];
#pragma unroll
  for (int f = 0; f < 4; ++f)
#pragma unroll
    for (int r = 0; r < 8; ++r) accO[f].f[r] = 0.0f;
  __syncthreads();

  const float scale = 0.08838834764831845f;  // 1/sqrt(128)
  const float slope = exp2f(-0.5f * (float)(h + 1));
  constexpr float LOG2E = 1.4426950408889634f;
  const int srow = tid >> 2;                 // softmax row owned by this thread
  const int part = tid & 3;
  const int trow = it * 64 + srow;           // query position t

  for (int j = 0; j <= it; ++j) {
    const int krow0 = bb * 2048 + j * 64;

    // --- async copy K chunk (64x128 f16) to LDS ---
#pragma unroll
    for (int i = 0; i < 4; ++i) {
      const int s = tid + (i << 8);
      const int r = s >> 4, c = (s & 15) << 3;
      const uint32_t ldsoff = (uint32_t)(uintptr_t)(&Ks[r][c]);
      const uint64_t ga = (uint64_t)(uintptr_t)(QKV + (size_t)(krow0 + r) * 3072 + kcol + c);
      asm volatile("global_load_async_to_lds_b128 %0, %1, off"
                   :: "v"(ldsoff), "v"(ga) : "memory");
    }
    asm volatile("s_wait_asynccnt 0x0" ::: "memory");
    __syncthreads();

    // --- S = Q K^T for this 64x64 chunk ---
    const int smo = (wid & 3) << 4, sno = (wid >> 2) << 5;
    FragF s0, s1;
#pragma unroll
    for (int r = 0; r < 8; ++r) { s0.f[r] = 0.0f; s1.f[r] = 0.0f; }
#pragma unroll
    for (int d0 = 0; d0 < 128; d0 += 32) {
      FragH a, b0, b1;
      load_a(a,  &Qs[0][d0], 130, smo);
      load_b(b0, &Ks[0][d0], 130, sno);
      load_b(b1, &Ks[0][d0], 130, sno + 16);
      s0.v = wmma16(a, b0, s0.v);
      s1.v = wmma16(a, b1, s1.v);
    }
#pragma unroll
    for (int r = 0; r < 8; ++r) {
      const int m = smo + ((lane >> 4) << 3) + r;
      const int n = lane & 15;
      Sf[m][sno + n]      = s0.f[r];
      Sf[m][sno + 16 + n] = s1.f[r];
    }
    __syncthreads();

    // --- online softmax: 4 threads per row, 16 cols each ---
    float vals[16];
    float mx = -1e30f;
#pragma unroll
    for (int c16 = 0; c16 < 16; ++c16) {
      const int c = (part << 4) + c16;
      const int key = j * 64 + c;
      float sv = Sf[srow][c] * scale + slope * (float)(key - trow);
      if (key > trow) sv = -1e30f;           // causal mask
      vals[c16] = sv;
      mx = fmaxf(mx, sv);
    }
    mx = fmaxf(mx, __shfl_xor(mx, 1));
    mx = fmaxf(mx, __shfl_xor(mx, 2));
    const float m_old = row_m[srow];
    const float m_new = fmaxf(m_old, mx);
    const float csc = exp2f((m_old - m_new) * LOG2E);
    if (part == 0) { row_m[srow] = m_new; row_scale[srow] = csc; }
    __syncthreads();                          // everyone done reading Sf

    // --- async copy V chunk into Vs (aliases Sf) ---
#pragma unroll
    for (int i = 0; i < 4; ++i) {
      const int s = tid + (i << 8);
      const int r = s >> 4, c = (s & 15) << 3;
      const uint32_t ldsoff = (uint32_t)(uintptr_t)(&Vs[r][c]);
      const uint64_t ga = (uint64_t)(uintptr_t)(QKV + (size_t)(krow0 + r) * 3072 + vcol + c);
      asm volatile("global_load_async_to_lds_b128 %0, %1, off"
                   :: "v"(ldsoff), "v"(ga) : "memory");
    }

    // --- p = exp(s - m_new), write P (f16), accumulate row sum ---
    float psum = 0.0f;
#pragma unroll
    for (int c16 = 0; c16 < 16; ++c16) {
      const float p = exp2f((vals[c16] - m_new) * LOG2E);
      psum += p;
      Ps[srow][(part << 4) + c16] = (_Float16)p;
    }
    psum += __shfl_xor(psum, 1);
    psum += __shfl_xor(psum, 2);
    if (part == 0) row_l[srow] = row_l[srow] * csc + psum;

    asm volatile("s_wait_asynccnt 0x0" ::: "memory");
    __syncthreads();                          // Ps, Vs, row_scale visible

    // --- rescale O accumulators, then O += P @ V ---
    const int pmo = (wid & 3) << 4, pno = (wid >> 2) << 6;
#pragma unroll
    for (int f = 0; f < 4; ++f)
#pragma unroll
      for (int r = 0; r < 8; ++r) {
        const int m = pmo + ((lane >> 4) << 3) + r;
        accO[f].f[r] *= row_scale[m];
      }
#pragma unroll
    for (int kk = 0; kk < 64; kk += 32) {
      FragH a;
      load_a(a, &Ps[0][kk], 66, pmo);
#pragma unroll
      for (int f = 0; f < 4; ++f) {
        FragH bv;
        load_bv(bv, &Vs[0][0], 130, kk, pno + (f << 4));
        accO[f].v = wmma16(a, bv, accO[f].v);
      }
    }
    __syncthreads();                          // protect Ks/Sf/Ps for next chunk
  }

  // --- normalize and store O (f16) ---
  const int pmo = (wid & 3) << 4, pno = (wid >> 2) << 6;
#pragma unroll
  for (int f = 0; f < 4; ++f)
#pragma unroll
    for (int r = 0; r < 8; ++r) {
      const int m = pmo + ((lane >> 4) << 3) + r;
      const float o = accO[f].f[r] / row_l[m];
      const int n = pno + (f << 4) + (lane & 15);
      OF[(size_t)(row0 + m) * 2048 + qcol + n] = (_Float16)o;
    }
}

// ---------------------------------------------------------------------------
// K3: out[4096][2048] (f32) = of[4096][2048] (f16) @ proj_w[2048][2048]^T + b
// Same tiling as K1; A is already f16.
// ---------------------------------------------------------------------------
__global__ __launch_bounds__(256, 1)
void k_proj_gemm(const _Float16* __restrict__ A, const float* __restrict__ W,
                 const float* __restrict__ BIAS, float* __restrict__ OUT) {
  constexpr int TM = 128, TN = 128, TK = 32, LDT = TK + 2;
  __shared__ _Float16 As[TM][LDT];
  __shared__ _Float16 Bs[TN][LDT];

  const int tid = threadIdx.x;
  const int wid = tid >> 5;
  const int lane = tid & 31;
  const int bm = blockIdx.x & 31;          // 32 M-blocks
  const int bn = blockIdx.x >> 5;          // 16 N-blocks
  const int m0 = bm * TM, n0 = bn * TN;
  const int wm = (wid & 3) * 32;
  const int wn = (wid >> 2) * 64;

  FragF acc[2][4];
#pragma unroll
  for (int i = 0; i < 2; ++i)
#pragma unroll
    for (int j = 0; j < 4; ++j)
#pragma unroll
      for (int r = 0; r < 8; ++r) acc[i][j].f[r] = 0.0f;

  // A: 128 rows x 4 uint4 = 512 slots -> 2/thread; B: 128 rows x 8 float4 -> 4/thread
  uint4 qa[2];
  float4 pb[4];
#pragma unroll
  for (int i = 0; i < 2; ++i) {
    const int s = tid + (i << 8);
    const int row = s >> 2, cp = (s & 3) << 3;
    qa[i] = *(const uint4*)(A + (size_t)(m0 + row) * 2048 + cp);
  }
#pragma unroll
  for (int i = 0; i < 4; ++i) {
    const int s = tid + (i << 8);
    const int row = s >> 3, kp = (s & 7) << 2;
    pb[i] = *(const float4*)(W + (size_t)(n0 + row) * 2048 + kp);
  }
#pragma unroll
  for (int i = 0; i < 2; ++i) {
    const int s = tid + (i << 8);
    const int row = s >> 2, cp = (s & 3) << 3;
    *(uint32_t*)&As[row][cp]     = qa[i].x;
    *(uint32_t*)&As[row][cp + 2] = qa[i].y;
    *(uint32_t*)&As[row][cp + 4] = qa[i].z;
    *(uint32_t*)&As[row][cp + 6] = qa[i].w;
  }
#pragma unroll
  for (int i = 0; i < 4; ++i) {
    const int s = tid + (i << 8);
    const int row = s >> 3, kp = (s & 7) << 2;
    *(uint32_t*)&Bs[row][kp]     = pack_f2(pb[i].x, pb[i].y);
    *(uint32_t*)&Bs[row][kp + 2] = pack_f2(pb[i].z, pb[i].w);
  }
  __syncthreads();

  for (int k0 = TK; k0 <= 2048; k0 += TK) {
    const bool has_next = (k0 < 2048);
    if (has_next) {
#pragma unroll
      for (int i = 0; i < 2; ++i) {
        const int s = tid + (i << 8);
        const int row = s >> 2, cp = (s & 3) << 3;
        qa[i] = *(const uint4*)(A + (size_t)(m0 + row) * 2048 + k0 + cp);
      }
#pragma unroll
      for (int i = 0; i < 4; ++i) {
        const int s = tid + (i << 8);
        const int row = s >> 3, kp = (s & 7) << 2;
        pb[i] = *(const float4*)(W + (size_t)(n0 + row) * 2048 + k0 + kp);
      }
    }
    FragH a0, a1, b[4];
    load_a(a0, &As[0][0], LDT, wm);
    load_a(a1, &As[0][0], LDT, wm + 16);
#pragma unroll
    for (int j = 0; j < 4; ++j) load_b(b[j], &Bs[0][0], LDT, wn + (j << 4));
#pragma unroll
    for (int j = 0; j < 4; ++j) {
      acc[0][j].v = wmma16(a0, b[j], acc[0][j].v);
      acc[1][j].v = wmma16(a1, b[j], acc[1][j].v);
    }
    if (has_next) {
      __syncthreads();
#pragma unroll
      for (int i = 0; i < 2; ++i) {
        const int s = tid + (i << 8);
        const int row = s >> 2, cp = (s & 3) << 3;
        *(uint32_t*)&As[row][cp]     = qa[i].x;
        *(uint32_t*)&As[row][cp + 2] = qa[i].y;
        *(uint32_t*)&As[row][cp + 4] = qa[i].z;
        *(uint32_t*)&As[row][cp + 6] = qa[i].w;
      }
#pragma unroll
      for (int i = 0; i < 4; ++i) {
        const int s = tid + (i << 8);
        const int row = s >> 3, kp = (s & 7) << 2;
        *(uint32_t*)&Bs[row][kp]     = pack_f2(pb[i].x, pb[i].y);
        *(uint32_t*)&Bs[row][kp + 2] = pack_f2(pb[i].z, pb[i].w);
      }
      __syncthreads();
    }
  }

#pragma unroll
  for (int i = 0; i < 2; ++i)
#pragma unroll
    for (int j = 0; j < 4; ++j)
#pragma unroll
      for (int r = 0; r < 8; ++r) {
        const int ml = wm + (i << 4) + ((lane >> 4) << 3) + r;
        const int nl = wn + (j << 4) + (lane & 15);
        OUT[(size_t)(m0 + ml) * 2048 + (n0 + nl)] = acc[i][j].f[r] + BIAS[n0 + nl];
      }
}

// ---------------------------------------------------------------------------
extern "C" void kernel_launch(void* const* d_in, const int* in_sizes, int n_in,
                              void* d_out, int out_size, void* d_ws, size_t ws_size,
                              hipStream_t stream) {
  (void)in_sizes; (void)n_in; (void)out_size; (void)ws_size;
  const float* x      = (const float*)d_in[0];
  // d_in[1] = attn_mask (causal, computed analytically)
  // d_in[2] = alibi_bias (computed analytically from head index)
  const float* qkv_w  = (const float*)d_in[3];
  const float* qkv_b  = (const float*)d_in[4];
  const float* proj_w = (const float*)d_in[5];
  const float* proj_b = (const float*)d_in[6];

  _Float16* qkvh = (_Float16*)d_ws;                                    // [4096][3072] f16 (24 MB)
  _Float16* of   = (_Float16*)((char*)d_ws + (size_t)4096 * 3072 * 2); // [4096][2048] f16 (16 MB)
  float* out = (float*)d_out;

  k_qkv_gemm<<<dim3(32 * 24), 256, 0, stream>>>(x, qkv_w, qkv_b, qkvh);
  k_attn<<<dim3(2 * 16 * 32), 256, 0, stream>>>(qkvh, of);
  k_proj_gemm<<<dim3(32 * 16), 256, 0, stream>>>(of, proj_w, proj_b, out);
}
